// TemporalGCN_19232863551793
// MI455X (gfx1250) — compile-verified, matrix operand order checked by reading, not run on hardware
//
#include <hip/hip_runtime.h>

typedef __attribute__((ext_vector_type(16))) __bf16    v16bf;
typedef __attribute__((ext_vector_type(8)))  float     v8f;
typedef __attribute__((ext_vector_type(4)))  unsigned  v4u;
typedef __attribute__((ext_vector_type(8)))  int       v8i;
typedef __attribute__((ext_vector_type(4)))  int       v4i;

#define HID 128
#define TSTEPS 2

// ---------------------------------------------------------------------------
// B-fragment swizzle for v_wmma_f32_16x16x32_bf16:
// W(k,n) row-major -> [kt(4)][nt(8)][lane(32)][j(16)] so each lane's 16-bf16
// fragment is 32 contiguous bytes (two ds_load_b128).
// ---------------------------------------------------------------------------
__device__ __forceinline__ int wsw_index(int k, int n) {
    int kt = k >> 5, kk = k & 31;
    int nt = n >> 4, nn = n & 15;
    int grp = (kk >> 3) & 1;                   // which 16-lane half holds this K
    int j   = (kk & 7) | ((kk >> 4) << 3);     // slot within the 16-elem fragment
    int lane = nn + (grp << 4);
    return (((kt << 3) + nt) << 9) + (lane << 4) + j;
}

// One-time: convert W f32 -> bf16 and pre-swizzle into fragment order (32KB).
__global__ void prep_w_kernel(const float* __restrict__ W, __bf16* __restrict__ Wsw) {
    int i = blockIdx.x * blockDim.x + threadIdx.x;
    if (i >= HID * HID) return;
    Wsw[wsw_index(i >> 7, i & 127)] = (__bf16)W[i];
}

// ---------------------------------------------------------------------------
// Simple fills
// ---------------------------------------------------------------------------
__global__ void fill_f32(float* p, float v, int n) {
    int i = blockIdx.x * blockDim.x + threadIdx.x;
    if (i < n) p[i] = v;
}

__global__ void fill_f32x4(float4* p, float v, int n4) {
    int i = blockIdx.x * blockDim.x + threadIdx.x;
    if (i < n4) p[i] = make_float4(v, v, v, v);
}

// ---------------------------------------------------------------------------
// Degree: deg pre-initialized to 1.0 (self loop); +1 per incoming edge
// ---------------------------------------------------------------------------
__global__ void deg_kernel(const int* __restrict__ dst, float* __restrict__ deg, int E) {
    int e = blockIdx.x * blockDim.x + threadIdx.x;
    if (e < E) atomicAdd(&deg[dst[e]], 1.0f);
}

__global__ void rsqrt_kernel(float* __restrict__ deg, int n) {
    int i = blockIdx.x * blockDim.x + threadIdx.x;
    if (i < n) deg[i] = rsqrtf(deg[i]);   // deg >= 1 always
}

// ---------------------------------------------------------------------------
// GEMM: Y[N,128] = X[N,128] @ W[128,128]  via v_wmma_f32_16x16x32_bf16
//  - pre-swizzled bf16 W (32KB) DMA'd to LDS by the Tensor Data Mover
//  - 8 waves / block, one wave = 16 rows x 128 cols, 32 WMMAs per wave
// ---------------------------------------------------------------------------
__global__ void gemm_wmma_kernel(const float* __restrict__ X,
                                 const __bf16* __restrict__ Wsw,
                                 float* __restrict__ Y, int nrows) {
    __shared__ __bf16 ldsW[4 * 8 * 32 * 16];   // 32KB

#if defined(__gfx1250__) && __has_builtin(__builtin_amdgcn_tensor_load_to_lds)
    // --- TDM: one per-wave DMA of the whole 32KB tile (wave 0 only). ---
    if (threadIdx.x < 32) {
        unsigned ldsAddr = (unsigned)(size_t)&ldsW[0];          // low 32 bits = LDS offset
        unsigned long long ga = (unsigned long long)(size_t)Wsw;

        // D# group 0: count=1 | lds_addr | global_addr[56:0] | type=2
        v4u g0;
        g0.x = 1u;                                              // count=1 (valid user D#)
        g0.y = ldsAddr;                                         // bits 63:32
        g0.z = (unsigned)(ga & 0xffffffffu);                    // global_addr[31:0]
        g0.w = (unsigned)((ga >> 32) & 0x01ffffffu) | (2u << 30); // [56:32] | type=2

        // D# group 1: data_size=1 (2B), 1-D tensor_dim0=tile_dim0=16384
        v8i g1;
        g1[0] = 0x00010000;   // workgroup_mask=0 | data_size=1 (2 bytes)
        g1[1] = 0x40000000;   // tensor_dim0[15:0]=0x4000 at bits 63:48
        g1[2] = 0x00010000;   // tensor_dim0[31:16]=0 | tensor_dim1=1 at bits 95:80
        g1[3] = 0x40000000;   // tensor_dim1[31:16]=0 | tile_dim0=0x4000 at bits 127:112
        g1[4] = 0;            // tile_dim1=0 (1-D), tile_dim2=0
        g1[5] = 0x00004000;   // tensor_dim0_stride = 16384
        g1[6] = 0;
        g1[7] = 0;

        v4i gz = {0, 0, 0, 0};
#if __clang_major__ >= 23
        v8i gz8 = {0, 0, 0, 0, 0, 0, 0, 0};
        __builtin_amdgcn_tensor_load_to_lds(g0, g1, gz, gz, gz8, 0);
#else
        __builtin_amdgcn_tensor_load_to_lds(g0, g1, gz, gz, 0);
#endif
        __builtin_amdgcn_s_wait_tensorcnt((short)0);
    }
    __syncthreads();
#else
    // Fallback: cooperative plain copy of the pre-swizzled 32KB.
    {
        const uint4* sw = (const uint4*)Wsw;
        uint4*       dw = (uint4*)ldsW;
        for (int i = threadIdx.x; i < 2048; i += blockDim.x) dw[i] = sw[i];
    }
    __syncthreads();
#endif

    const int wave = threadIdx.x >> 5;
    const int lane = threadIdx.x & 31;
    const int sub  = lane & 15;
    const int grp  = lane >> 4;

    const int rowTile = blockIdx.x * 8 + wave;
    const int m0 = rowTile * 16;
    if (m0 >= nrows) return;

    int arow = m0 + sub;
    if (arow >= nrows) arow = nrows - 1;            // clamp (guarded at store)
    const float* xrow = X + (size_t)arow * HID;

    // Burst-load the full 128-float row (16 x b128) before the K loop.
    float4 p[16];
#pragma unroll
    for (int kt = 0; kt < 4; ++kt) {
        const int kb = kt * 32 + grp * 8;
        p[kt * 4 + 0] = *(const float4*)(xrow + kb);
        p[kt * 4 + 1] = *(const float4*)(xrow + kb + 4);
        p[kt * 4 + 2] = *(const float4*)(xrow + kb + 16);
        p[kt * 4 + 3] = *(const float4*)(xrow + kb + 20);
    }

    v8f acc[8] = {};

#pragma unroll
    for (int kt = 0; kt < 4; ++kt) {
        float4 p0 = p[kt * 4 + 0], p1 = p[kt * 4 + 1];
        float4 p2 = p[kt * 4 + 2], p3 = p[kt * 4 + 3];

        v16bf a;
        a[0]  = (__bf16)p0.x; a[1]  = (__bf16)p0.y; a[2]  = (__bf16)p0.z; a[3]  = (__bf16)p0.w;
        a[4]  = (__bf16)p1.x; a[5]  = (__bf16)p1.y; a[6]  = (__bf16)p1.z; a[7]  = (__bf16)p1.w;
        a[8]  = (__bf16)p2.x; a[9]  = (__bf16)p2.y; a[10] = (__bf16)p2.z; a[11] = (__bf16)p2.w;
        a[12] = (__bf16)p3.x; a[13] = (__bf16)p3.y; a[14] = (__bf16)p3.z; a[15] = (__bf16)p3.w;

#pragma unroll
        for (int nt = 0; nt < 8; ++nt) {
            const v16bf b = *(const v16bf*)&ldsW[(((kt << 3) + nt) << 9) + (lane << 4)];
            acc[nt] = __builtin_amdgcn_wmma_f32_16x16x32_bf16(
                false, a, false, b, (short)0, acc[nt], false, false);
        }
    }

    // C/D layout: VGPR r -> row m0 + grp*8 + r, col = nt*16 + (lane&15)
#pragma unroll
    for (int nt = 0; nt < 8; ++nt) {
        float* yp = Y + (size_t)(m0 + grp * 8) * HID + nt * 16 + sub;
#pragma unroll
        for (int r = 0; r < 8; ++r) {
            int orow = m0 + grp * 8 + r;
            if (orow < nrows) yp[(size_t)r * HID] = acc[nt][r];
        }
    }
}

// ---------------------------------------------------------------------------
// Edge scatter: one wave per edge; lane handles one float4 of the 128-vector.
// out[dst] += h[src] * dinv[src]*dinv[dst]   (f32 atomics; rows live in L2)
// ---------------------------------------------------------------------------
__global__ void agg_kernel(const int* __restrict__ src, const int* __restrict__ dst,
                           const float* __restrict__ h, const float* __restrict__ dinv,
                           float* __restrict__ out, int E) {
    int gid  = blockIdx.x * blockDim.x + threadIdx.x;
    int e    = gid >> 5;
    int lane = gid & 31;
    if (e >= E) return;

    int s = src[e];
    int d = dst[e];
    float norm = dinv[s] * dinv[d];

    float* o = out + (size_t)d * HID + lane * 4;
    __builtin_prefetch(o, 1, 0);   // warm destination lines ahead of atomics

    float4 v = ((const float4*)(h + (size_t)s * HID))[lane];
    atomicAdd(o + 0, v.x * norm);
    atomicAdd(o + 1, v.y * norm);
    atomicAdd(o + 2, v.z * norm);
    atomicAdd(o + 3, v.w * norm);
}

// ---------------------------------------------------------------------------
// Fused self-loop + bias (+ optional ReLU):
// dst = [relu](agg + lin*dinv[node]^2 + bias)
// ---------------------------------------------------------------------------
__global__ void fuse_kernel(const float4* __restrict__ agg, const float4* __restrict__ lin,
                            const float* __restrict__ dinv, const float4* __restrict__ bias,
                            float4* __restrict__ dst, int n32, int do_relu) {
    int g = blockIdx.x * blockDim.x + threadIdx.x;
    if (g >= n32) return;
    int node = g >> 5;
    int c    = g & 31;
    float di = dinv[node];
    float s  = di * di;
    float4 a = agg[g], l = lin[g], b = bias[c];
    float4 r;
    r.x = fmaf(l.x, s, a.x) + b.x;
    r.y = fmaf(l.y, s, a.y) + b.y;
    r.z = fmaf(l.z, s, a.z) + b.z;
    r.w = fmaf(l.w, s, a.w) + b.w;
    if (do_relu) {
        r.x = fmaxf(r.x, 0.f); r.y = fmaxf(r.y, 0.f);
        r.z = fmaxf(r.z, 0.f); r.w = fmaxf(r.w, 0.f);
    }
    dst[g] = r;
}

// ---------------------------------------------------------------------------
// importance[i] = dot(final[i,:], Wc) + bc
// ---------------------------------------------------------------------------
__global__ void importance_kernel(const float4* __restrict__ fin, const float4* __restrict__ wc,
                                  const float* __restrict__ bc, float* __restrict__ out, int n) {
    int i = blockIdx.x * blockDim.x + threadIdx.x;
    if (i >= n) return;
    float acc = 0.f;
#pragma unroll
    for (int j = 0; j < 32; ++j) {
        float4 f = fin[(size_t)i * 32 + j];
        float4 w = wc[j];
        acc += f.x * w.x + f.y * w.y + f.z * w.z + f.w * w.w;
    }
    out[i] = acc + bc[0];
}

// ---------------------------------------------------------------------------
// Launch
// ---------------------------------------------------------------------------
extern "C" void kernel_launch(void* const* d_in, const int* in_sizes, int n_in,
                              void* d_out, int out_size, void* d_ws, size_t ws_size,
                              hipStream_t stream) {
    const float* x_seq = (const float*)d_in[0];   // [T,N,128] f32
    const int*   eidx  = (const int*)  d_in[1];   // [T,2,E]
    const float* W1    = (const float*)d_in[2];
    const float* b1    = (const float*)d_in[3];
    const float* W2    = (const float*)d_in[4];
    const float* b2    = (const float*)d_in[5];
    const float* Wc    = (const float*)d_in[6];
    const float* bc    = (const float*)d_in[7];

    const int N = in_sizes[0] / (TSTEPS * HID);
    const int E = in_sizes[1] / (TSTEPS * 2);

    float* out_imp = (float*)d_out;                 // [N]
    float* out_t0  = out_imp + N;                   // [N,128]
    float* out_t1  = out_t0 + (size_t)N * HID;      // [N,128]

    float*  ws     = (float*)d_ws;
    float*  dinv   = ws;                            // N floats
    float*  bufLin = ws + (size_t)((N + 255) & ~255);
    float*  bufAgg = bufLin + (size_t)N * HID;
    __bf16* wsw1   = (__bf16*)(bufAgg + (size_t)N * HID);   // 32KB
    __bf16* wsw2   = wsw1 + HID * HID;                      // 32KB

    const int BLK = 256;
    const int n32      = N * 32;                    // float4 elements per feature matrix
    const int nbN      = (N + BLK - 1) / BLK;
    const int nbN4     = (n32 + BLK - 1) / BLK;
    const int nbE      = (E + BLK - 1) / BLK;
    const int tiles    = (N + 15) / 16;
    const int nbGemm   = (tiles + 7) / 8;
    const long eth     = (long)E * 32;
    const int nbAgg    = (int)((eth + BLK - 1) / BLK);
    const int nbW      = (HID * HID + BLK - 1) / BLK;

    // One-time: weights -> bf16, pre-swizzled into WMMA B-fragment order.
    prep_w_kernel<<<nbW, BLK, 0, stream>>>(W1, wsw1);
    prep_w_kernel<<<nbW, BLK, 0, stream>>>(W2, wsw2);

    for (int t = 0; t < TSTEPS; ++t) {
        const float* x_t = x_seq + (size_t)t * N * HID;
        const int*   src = eidx + (size_t)t * 2 * E;
        const int*   dst = src + E;

        // symmetric-normalization coefficients
        fill_f32<<<nbN, BLK, 0, stream>>>(dinv, 1.0f, N);          // self loop
        deg_kernel<<<nbE, BLK, 0, stream>>>(dst, dinv, E);
        rsqrt_kernel<<<nbN, BLK, 0, stream>>>(dinv, N);

        // conv1: linear -> scatter -> self+bias+relu
        gemm_wmma_kernel<<<nbGemm, BLK, 0, stream>>>(x_t, wsw1, bufLin, N);
        fill_f32x4<<<nbN4, BLK, 0, stream>>>((float4*)bufAgg, 0.0f, n32);
        agg_kernel<<<nbAgg, BLK, 0, stream>>>(src, dst, bufLin, dinv, bufAgg, E);
        fuse_kernel<<<nbN4, BLK, 0, stream>>>((const float4*)bufAgg, (const float4*)bufLin,
                                              dinv, (const float4*)b1, (float4*)bufLin,
                                              n32, 1);

        // conv2: in-place linear (row-wise GEMM: each wave stores only rows it
        // has fully consumed) -> scatter -> self+bias (no relu) into d_out
        gemm_wmma_kernel<<<nbGemm, BLK, 0, stream>>>(bufLin, wsw2, bufLin, N);
        fill_f32x4<<<nbN4, BLK, 0, stream>>>((float4*)bufAgg, 0.0f, n32);
        agg_kernel<<<nbAgg, BLK, 0, stream>>>(src, dst, bufLin, dinv, bufAgg, E);

        float* outT = (t == 0) ? out_t0 : out_t1;
        fuse_kernel<<<nbN4, BLK, 0, stream>>>((const float4*)bufAgg, (const float4*)bufLin,
                                              dinv, (const float4*)b2, (float4*)outT,
                                              n32, 0);
    }

    // final projection from last timestep output
    importance_kernel<<<nbN, BLK, 0, stream>>>((const float4*)out_t1, (const float4*)Wc,
                                               bc, out_imp, N);
}
